// SlotAttention_8701603742048
// MI455X (gfx1250) — matrix operand order repaired and checked by reading, not compiled, not run on hardware
//
#include <hip/hip_runtime.h>
#include <hip/hip_fp16.h>
#include <float.h>

// Slot attention, MI455X (gfx1250, wave32, WMMA f16->f32).
// B=64, N=4096, D=256, S=8, ITERS=3.
// All matrix fragments are loaded/stored as contiguous 32B/16B vectors:
//  - weights pre-packed into WMMA-B fragment order (prep_kernel)
//  - v stored in packed-B fragment order straight out of the k/v GEMM
//  - layernorm output written to LDS in packed-A fragment order (b128 stores)

#define Bsz 64
#define Nn  4096
#define Dd  256
#define Ss  8
#define NCH (Nn / 32)   // 128 n-chunks of 32

typedef __attribute__((ext_vector_type(16))) _Float16 v16h;
typedef __attribute__((ext_vector_type(8)))  _Float16 v8h;
typedef __attribute__((ext_vector_type(8)))  float    v8f;

__device__ __forceinline__ v8f wmma16(v16h a, v16h b, v8f c) {
  return __builtin_amdgcn_wmma_f32_16x16x32_f16(false, a, false, b, (short)0, c,
                                                false, false);
}

__device__ __forceinline__ float wave_sum(float v) {
#pragma unroll
  for (int m = 16; m >= 1; m >>= 1) v += __shfl_xor(v, m, 32);
  return v;
}

// ---- packed fragment layouts ----------------------------------------------
// A (16x32 f16): lane = m + 16*(k'>=16),     idx = k'&15
// B (32x16 f16): lane = n + 16*(k'>=16),     idx = k'&15
// Packed buffers store, per 32-K chunk, 32 lanes x 16 f16 contiguously, so a
// fragment is one v16h (2 x b128) load at (chunk*32 + lane)*16.

__device__ __forceinline__ v16h load_frag(const _Float16* p, int chunk,
                                          int lane) {
  return *(const v16h*)(p + ((size_t)(chunk * 32 + lane)) * 16);
}

// k^T fragment straight from row-major k[n][d]: per lane 16 contiguous f16.
__device__ __forceinline__ v16h load_frag_kT(const _Float16* kb, int ncol0,
                                             int kd0, int lane) {
  const int n = ncol0 + (lane & 15);
  return *(const v16h*)(kb + (size_t)n * Dd + kd0 + ((lane >> 4) << 4));
}

// One wave layernorms one 256-elem row and writes it into a packed-A LDS
// buffer (8 chunks x 32 lanes x 16 f16) as a single b128 store per lane.
__device__ __forceinline__ void wave_ln_row_packA(
    const float* __restrict__ xp, const float* __restrict__ g,
    const float* __restrict__ bb, _Float16* __restrict__ apk, int m, int lane) {
  const int c0 = lane * 8;
  const float4 x0 = *(const float4*)(xp + c0);
  const float4 x1 = *(const float4*)(xp + c0 + 4);
  float xv[8] = {x0.x, x0.y, x0.z, x0.w, x1.x, x1.y, x1.z, x1.w};
  float s = 0.0f, s2 = 0.0f;
#pragma unroll
  for (int j = 0; j < 8; ++j) { s += xv[j]; s2 += xv[j] * xv[j]; }
  s = wave_sum(s);
  s2 = wave_sum(s2);
  const float mu   = s * (1.0f / 256.0f);
  const float var  = s2 * (1.0f / 256.0f) - mu * mu;
  const float rstd = rsqrtf(var + 1e-5f);
  const float4 g0 = *(const float4*)(g + c0);
  const float4 g1 = *(const float4*)(g + c0 + 4);
  const float4 b0 = *(const float4*)(bb + c0);
  const float4 b1 = *(const float4*)(bb + c0 + 4);
  const float gv[8] = {g0.x, g0.y, g0.z, g0.w, g1.x, g1.y, g1.z, g1.w};
  const float bv[8] = {b0.x, b0.y, b0.z, b0.w, b1.x, b1.y, b1.z, b1.w};
  v8h out;
#pragma unroll
  for (int j = 0; j < 8; ++j)
    out[j] = (_Float16)((xv[j] - mu) * rstd * gv[j] + bv[j]);
  // c = lane*8 + j  ->  chunk = lane>>2, lane' = m + 16*((lane>>1)&1),
  //                     idx = 8*(lane&1) + j   (j contiguous)
  const int kc = lane >> 2;
  const int lp = m + (((lane >> 1) & 1) << 4);
  const int ib = (lane & 1) << 3;
  *(v8h*)(apk + (size_t)(kc * 32 + lp) * 16 + ib) = out;
}

// ---- Kernel 0: pack weights (B-fragment order) + slots init ----------------
__global__ __launch_bounds__(256) void prep_kernel(
    const float* __restrict__ Wq, const float* __restrict__ Wk,
    const float* __restrict__ Wv, const float* __restrict__ W1,
    const float* __restrict__ W2, const float* __restrict__ slots_init,
    _Float16* Wq_p, _Float16* Wk_p, _Float16* Wv_p, _Float16* W1_p,
    _Float16* W2_p, float* slots) {
  const int idx = blockIdx.x * 256 + threadIdx.x;
  const int WSZ = Dd * Dd;
  const int w = idx / WSZ;         // which weight
  const int e = idx - w * WSZ;     // element within it
  const int k = e >> 8;            // row (K dim)
  const int n = e & 255;           // col (N dim)
  const size_t dst = ((size_t)((k >> 5) * 16 + (n >> 4)) * 32 +
                      (n & 15) + (((k >> 4) & 1) << 4)) * 16 + (k & 15);
  const float* src[5] = {Wq, Wk, Wv, W1, W2};
  _Float16* dstp[5]   = {Wq_p, Wk_p, Wv_p, W1_p, W2_p};
  if (w < 5) dstp[w][dst] = (_Float16)src[w][e];
  if (idx < Bsz * Ss * Dd) slots[idx] = slots_init[idx & (Ss * Dd - 1)];
}

// ---- Kernel 1: fused LN(inputs) + K/V projection (dominant GEMM) -----------
__global__ __launch_bounds__(256) void ln_kv_kernel(
    const float* __restrict__ inp, const float* __restrict__ g_in,
    const float* __restrict__ b_in, const _Float16* __restrict__ Wk_p,
    const float* __restrict__ bk, const _Float16* __restrict__ Wv_p,
    const float* __restrict__ bv, _Float16* __restrict__ k_out,
    _Float16* __restrict__ v_p) {
  __shared__ _Float16 apk[8 * 32 * 16];   // packed-A x tile (8 KB)
  __shared__ _Float16 ksh[16][264];       // k staging for coalesced stores
  const int wave = threadIdx.x >> 5;
  const int lane = threadIdx.x & 31;
  const size_t row0 = (size_t)blockIdx.x * 16;
  const int b   = (int)(row0 >> 12);      // row0 / Nn
  const int rn0 = (int)(row0 & (Nn - 1)); // token row within batch
#pragma unroll
  for (int rr = 0; rr < 2; ++rr) {
    const int r = wave + rr * 8;
    wave_ln_row_packA(inp + (row0 + r) * Dd, g_in, b_in, apk, r, lane);
  }
  __syncthreads();

  // hoist all 8 A fragments (reused for both col tiles and both K/V)
  v16h afr[8];
#pragma unroll
  for (int kc = 0; kc < 8; ++kc) afr[kc] = load_frag(apk, kc, lane);

  const int vchunk  = rn0 >> 5;
  const int vhalf   = (rn0 >> 4) & 1;
  const int vlane   = (lane & 15) + (vhalf << 4);
#pragma unroll
  for (int t = 0; t < 2; ++t) {
    const int nt = wave * 2 + t;          // col tile 0..15
    const int n0 = nt * 16;
    v8f ck = {};
    v8f cv = {};
#pragma unroll
    for (int kc = 0; kc < 8; ++kc) {
      v16h bK = load_frag(Wk_p, kc * 16 + nt, lane);
      ck = wmma16(afr[kc], bK, ck);
      v16h bV = load_frag(Wv_p, kc * 16 + nt, lane);
      cv = wmma16(afr[kc], bV, cv);
    }
    const int n  = n0 + (lane & 15);
    const int mh = (lane >> 4) << 3;
    const float bkv = bk[n];
    const float bvv = bv[n];
    // k -> LDS (transposed store path)
#pragma unroll
    for (int r = 0; r < 8; ++r) ksh[mh + r][n] = (_Float16)(ck[r] + bkv);
    // v -> packed-B fragment order in global, one b128 store
    v8h vv;
#pragma unroll
    for (int r = 0; r < 8; ++r) vv[r] = (_Float16)(cv[r] + bvv);
    _Float16* vd = v_p + (((size_t)b * NCH + vchunk) * 16 + nt) * 512 +
                   (size_t)vlane * 16 + mh;
    *(v8h*)vd = vv;
  }
  __syncthreads();
  // coalesced k stores: each wave writes rows wave and wave+8
#pragma unroll
  for (int rr = 0; rr < 2; ++rr) {
    const int r = wave + rr * 8;
    const v8h kv = *(const v8h*)&ksh[r][lane * 8];
    *(v8h*)(k_out + (row0 + r) * Dd + lane * 8) = kv;
  }
}

// ---- Kernel 2: s = LN(slots); q = s@Wq + bq, written packed-A --------------
__global__ __launch_bounds__(256) void slot_q_kernel(
    const float* __restrict__ slots, const float* __restrict__ g_s,
    const float* __restrict__ b_s, const _Float16* __restrict__ Wq_p,
    const float* __restrict__ bq, _Float16* __restrict__ q_p) {
  __shared__ _Float16 apk[8 * 32 * 16];
  const int b = blockIdx.x;
  const int wave = threadIdx.x >> 5;
  const int lane = threadIdx.x & 31;
  // zero (covers padded slot rows 8..15), then LN rows 0..7
  *(v16h*)(apk + (size_t)threadIdx.x * 16) = (v16h)(_Float16)0.0f;
  __syncthreads();
  wave_ln_row_packA(slots + ((size_t)b * Ss + wave) * Dd, g_s, b_s, apk, wave,
                    lane);
  __syncthreads();
  v16h afr[8];
#pragma unroll
  for (int kc = 0; kc < 8; ++kc) afr[kc] = load_frag(apk, kc, lane);
  _Float16* qb = q_p + (size_t)b * 8 * 32 * 16;
#pragma unroll
  for (int t = 0; t < 2; ++t) {
    const int nt = wave * 2 + t;
    v8f c = {};
#pragma unroll
    for (int kc = 0; kc < 8; ++kc)
      c = wmma16(afr[kc], load_frag(Wq_p, kc * 16 + nt, lane), c);
    const int n = nt * 16 + (lane & 15);
    const float bias = bq[n];
    const int mh = (lane >> 4) << 3;
    // write q into packed-A fragment order (A of the dots GEMM): k = n
    const int kc   = n >> 5;
    const int lhi  = ((n >> 4) & 1) << 4;
    const int idx  = n & 15;
#pragma unroll
    for (int r = 0; r < 8; ++r)
      qb[(size_t)(kc * 32 + (mh + r) + lhi) * 16 + idx] =
          (_Float16)(c[r] + bias);
  }
}

// ---- Kernel 3: fused dots -> softmax(axis=S) -> N-norm -> updates ----------
__global__ __launch_bounds__(256) void attn_update_kernel(
    const _Float16* __restrict__ q_p, const _Float16* __restrict__ k_h,
    const _Float16* __restrict__ v_p, const unsigned char* __restrict__ mask,
    float* __restrict__ slots) {
  __shared__ _Float16 qsh[8 * 32 * 16];   // packed-A q (8 KB)
  __shared__ _Float16 ash[4 * 32 * 16];   // packed-A attn super-tile (4 KB)
  __shared__ float rsum[Ss];
  const int b    = blockIdx.x;
  const int wave = threadIdx.x >> 5;
  const int lane = threadIdx.x & 31;

  // stage q (already packed) with pure b128 copies
  *(v16h*)(qsh + (size_t)threadIdx.x * 16) =
      *(const v16h*)(q_p + (size_t)b * 4096 + (size_t)threadIdx.x * 16);
  if (threadIdx.x < Ss) rsum[threadIdx.x] = 0.0f;
  __syncthreads();

  v16h qfr[8];
#pragma unroll
  for (int kc = 0; kc < 8; ++kc) qfr[kc] = load_frag(qsh, kc, lane);

  const _Float16* kb = k_h + (size_t)b * Nn * Dd;
  const _Float16* vbase = v_p + (size_t)b * NCH * 16 * 512;
  const unsigned char* mb = mask + (size_t)b * Nn;
  const float scale = 0.0625f;  // D^-0.5

  v8f acc0 = {};
  v8f acc1 = {};
  float rs[8];
#pragma unroll
  for (int r = 0; r < 8; ++r) rs[r] = 0.0f;

  // attn LDS write coords (k = wave*16 + col is fixed per lane)
  const int col  = lane & 15;
  const int kk   = wave * 16 + col;
  const int akc  = kk >> 5;
  const int alhi = (wave & 1) << 4;
  const int aidx = col;

  for (int nt = 0; nt < Nn; nt += 128) {
    if (nt + 128 < Nn)
      __builtin_prefetch(kb + (size_t)(nt + 128 + wave * 16) * Dd, 0, 1);
    const int ncol0 = nt + wave * 16;
    // phase 1: dots tile for this wave's 16 columns
    v8f c = {};
#pragma unroll
    for (int kc = 0; kc < 8; ++kc)
      c = wmma16(qfr[kc], load_frag_kT(kb, ncol0, kc * 32, lane), c);
    const bool mk = mb[ncol0 + col] != 0;
    float mx = -FLT_MAX;
#pragma unroll
    for (int r = 0; r < 8; ++r) {
      const float dv = mk ? -FLT_MAX : c[r] * scale;  // masked -> uniform
      c[r] = dv;
      mx = fmaxf(mx, dv);
    }
    float se = 0.0f;
#pragma unroll
    for (int r = 0; r < 8; ++r) {
      const float e = __expf(c[r] - mx);
      c[r] = e;
      se += e;
    }
    const float inv = 1.0f / se;
    if (lane < 16) {  // valid slot rows 0..7
#pragma unroll
      for (int r = 0; r < 8; ++r) {
        const float av = c[r] * inv;
        ash[(size_t)(akc * 32 + r + alhi) * 16 + aidx] = (_Float16)av;
        rs[r] += av;
      }
    } else {          // zero padded rows 8..15
#pragma unroll
      for (int r = 0; r < 8; ++r)
        ash[(size_t)(akc * 32 + 8 + r + alhi) * 16 + aidx] = (_Float16)0.0f;
    }
    __syncthreads();
    // phase 2: updates += attn_tile @ v_tile (K = 128 n-columns)
#pragma unroll
    for (int kc = 0; kc < 4; ++kc) {
      const int chunk = (nt >> 5) + kc;
      v16h a = load_frag(ash, kc, lane);
      v16h b0 = *(const v16h*)(vbase + ((size_t)chunk * 16 + wave * 2) * 512 +
                               (size_t)lane * 16);
      acc0 = wmma16(a, b0, acc0);
      v16h b1 = *(const v16h*)(vbase + ((size_t)chunk * 16 + wave * 2 + 1) * 512 +
                               (size_t)lane * 16);
      acc1 = wmma16(a, b1, acc1);
    }
    __syncthreads();
  }
#pragma unroll
  for (int r = 0; r < 8; ++r) {
    const float v = wave_sum(rs[r]);  // lanes>=16 contributed 0
    if (lane == 0) atomicAdd(&rsum[r], v);
  }
  __syncthreads();
  if (lane < 16) {
    const int dcol = wave * 32 + lane;
#pragma unroll
    for (int r = 0; r < 8; ++r) {
      const float denom = rsum[r] + 1e-8f;
      float* sp = slots + ((size_t)b * Ss + r) * Dd;
      sp[dcol]      += acc0[r] / denom;
      sp[dcol + 16] += acc1[r] / denom;
    }
  }
}

// ---- Kernel 4: slots += relu(LN(slots)@W1 + b1) @ W2 + b2 ------------------
__global__ __launch_bounds__(256) void mlp_kernel(
    float* __restrict__ slots, const float* __restrict__ g_m,
    const float* __restrict__ b_m, const _Float16* __restrict__ W1_p,
    const float* __restrict__ b1, const _Float16* __restrict__ W2_p,
    const float* __restrict__ b2) {
  __shared__ _Float16 apkA[8 * 32 * 16];
  __shared__ _Float16 apkH[8 * 32 * 16];
  const int b = blockIdx.x;
  const int wave = threadIdx.x >> 5;
  const int lane = threadIdx.x & 31;
  *(v16h*)(apkA + (size_t)threadIdx.x * 16) = (v16h)(_Float16)0.0f;
  __syncthreads();
  wave_ln_row_packA(slots + ((size_t)b * Ss + wave) * Dd, g_m, b_m, apkA, wave,
                    lane);
  __syncthreads();
  v16h afr[8];
#pragma unroll
  for (int kc = 0; kc < 8; ++kc) afr[kc] = load_frag(apkA, kc, lane);
#pragma unroll
  for (int t = 0; t < 2; ++t) {
    const int nt = wave * 2 + t;
    v8f c = {};
#pragma unroll
    for (int kc = 0; kc < 8; ++kc)
      c = wmma16(afr[kc], load_frag(W1_p, kc * 16 + nt, lane), c);
    const int n = nt * 16 + (lane & 15);
    const float bias = b1[n];
    const int mh = (lane >> 4) << 3;
    // write h1 into packed-A order for GEMM2 (k = n); all 16 rows covered
    const int kc  = n >> 5;
    const int lhi = ((n >> 4) & 1) << 4;
    const int idx = n & 15;
#pragma unroll
    for (int r = 0; r < 8; ++r)
      apkH[(size_t)(kc * 32 + (mh + r) + lhi) * 16 + idx] =
          (_Float16)fmaxf(c[r] + bias, 0.0f);
  }
  __syncthreads();
  v16h hfr[8];
#pragma unroll
  for (int kc = 0; kc < 8; ++kc) hfr[kc] = load_frag(apkH, kc, lane);
#pragma unroll
  for (int t = 0; t < 2; ++t) {
    const int nt = wave * 2 + t;
    v8f c = {};
#pragma unroll
    for (int kc = 0; kc < 8; ++kc)
      c = wmma16(hfr[kc], load_frag(W2_p, kc * 16 + nt, lane), c);
    if (lane < 16) {  // valid slot rows 0..7 only
      const int n = nt * 16 + lane;
      const float bias = b2[n];
#pragma unroll
      for (int r = 0; r < 8; ++r)
        slots[((size_t)b * Ss + r) * Dd + n] += c[r] + bias;
    }
  }
}

// ---- Kernel 5: out = LN(slots, g_o, b_o), f32 ------------------------------
__global__ __launch_bounds__(256) void out_ln_kernel(
    const float* __restrict__ slots, const float* __restrict__ g_o,
    const float* __restrict__ b_o, float* __restrict__ out) {
  const int wave = threadIdx.x >> 5;
  const int lane = threadIdx.x & 31;
  const size_t row = (size_t)blockIdx.x * Ss + wave;
  const float* xp = slots + row * Dd;
  const int c0 = lane * 8;
  const float4 x0 = *(const float4*)(xp + c0);
  const float4 x1 = *(const float4*)(xp + c0 + 4);
  float xv[8] = {x0.x, x0.y, x0.z, x0.w, x1.x, x1.y, x1.z, x1.w};
  float s = 0.0f, s2 = 0.0f;
#pragma unroll
  for (int j = 0; j < 8; ++j) { s += xv[j]; s2 += xv[j] * xv[j]; }
  s = wave_sum(s);
  s2 = wave_sum(s2);
  const float mu   = s * (1.0f / 256.0f);
  const float var  = s2 * (1.0f / 256.0f) - mu * mu;
  const float rstd = rsqrtf(var + 1e-5f);
  const float4 g0 = *(const float4*)(g_o + c0);
  const float4 g1 = *(const float4*)(g_o + c0 + 4);
  const float4 b0 = *(const float4*)(b_o + c0);
  const float4 b1 = *(const float4*)(b_o + c0 + 4);
  const float gv[8] = {g0.x, g0.y, g0.z, g0.w, g1.x, g1.y, g1.z, g1.w};
  const float bv[8] = {b0.x, b0.y, b0.z, b0.w, b1.x, b1.y, b1.z, b1.w};
  float4 o0, o1;
  o0.x = (xv[0] - mu) * rstd * gv[0] + bv[0];
  o0.y = (xv[1] - mu) * rstd * gv[1] + bv[1];
  o0.z = (xv[2] - mu) * rstd * gv[2] + bv[2];
  o0.w = (xv[3] - mu) * rstd * gv[3] + bv[3];
  o1.x = (xv[4] - mu) * rstd * gv[4] + bv[4];
  o1.y = (xv[5] - mu) * rstd * gv[5] + bv[5];
  o1.z = (xv[6] - mu) * rstd * gv[6] + bv[6];
  o1.w = (xv[7] - mu) * rstd * gv[7] + bv[7];
  *(float4*)(out + row * Dd + c0) = o0;
  *(float4*)(out + row * Dd + c0 + 4) = o1;
}

extern "C" void kernel_launch(void* const* d_in, const int* in_sizes, int n_in,
                              void* d_out, int out_size, void* d_ws,
                              size_t ws_size, hipStream_t stream) {
  (void)in_sizes; (void)n_in; (void)out_size; (void)ws_size;
  const float* inputs      = (const float*)d_in[0];
  const unsigned char* msk = (const unsigned char*)d_in[1];  // bool [B,N]
  const float* slots_init  = (const float*)d_in[2];
  const float* g_in = (const float*)d_in[3];
  const float* b_in = (const float*)d_in[4];
  const float* g_s  = (const float*)d_in[5];
  const float* b_s  = (const float*)d_in[6];
  const float* g_m  = (const float*)d_in[7];
  const float* b_m  = (const float*)d_in[8];
  const float* g_o  = (const float*)d_in[9];
  const float* b_o  = (const float*)d_in[10];
  const float* Wq   = (const float*)d_in[11];
  const float* bq   = (const float*)d_in[12];
  const float* Wk   = (const float*)d_in[13];
  const float* bk   = (const float*)d_in[14];
  const float* Wv   = (const float*)d_in[15];
  const float* bv   = (const float*)d_in[16];
  const float* W1   = (const float*)d_in[17];
  const float* b1   = (const float*)d_in[18];
  const float* W2   = (const float*)d_in[19];
  const float* b2   = (const float*)d_in[20];

  char* ws = (char*)d_ws;
  size_t off = 0;
  _Float16* k_h  = (_Float16*)(ws + off); off += (size_t)Bsz * Nn * Dd * 2;
  _Float16* v_p  = (_Float16*)(ws + off); off += (size_t)Bsz * Nn * Dd * 2;
  _Float16* Wq_p = (_Float16*)(ws + off); off += (size_t)Dd * Dd * 2;
  _Float16* Wk_p = (_Float16*)(ws + off); off += (size_t)Dd * Dd * 2;
  _Float16* Wv_p = (_Float16*)(ws + off); off += (size_t)Dd * Dd * 2;
  _Float16* W1_p = (_Float16*)(ws + off); off += (size_t)Dd * Dd * 2;
  _Float16* W2_p = (_Float16*)(ws + off); off += (size_t)Dd * Dd * 2;
  _Float16* q_p  = (_Float16*)(ws + off); off += (size_t)Bsz * 16 * Dd * 2;
  float*    slots = (float*)(ws + off);   off += (size_t)Bsz * Ss * Dd * 4;

  prep_kernel<<<(5 * Dd * Dd) / 256, 256, 0, stream>>>(
      Wq, Wk, Wv, W1, W2, slots_init, Wq_p, Wk_p, Wv_p, W1_p, W2_p, slots);
  ln_kv_kernel<<<(Bsz * Nn) / 16, 256, 0, stream>>>(
      inputs, g_in, b_in, Wk_p, bk, Wv_p, bv, k_h, v_p);
  for (int it = 0; it < 3; ++it) {
    slot_q_kernel<<<Bsz, 256, 0, stream>>>(slots, g_s, b_s, Wq_p, bq, q_p);
    attn_update_kernel<<<Bsz, 256, 0, stream>>>(q_p, k_h, v_p, msk, slots);
    mlp_kernel<<<Bsz, 256, 0, stream>>>(slots, g_m, b_m, W1_p, b1, W2_p, b2);
  }
  out_ln_kernel<<<Bsz, 256, 0, stream>>>(slots, g_o, b_o, (float*)d_out);
}
